// HVBiLSTM_73315091742729
// MI455X (gfx1250) — matrix operand order repaired and checked by reading, not compiled, not run on hardware
//
#include <hip/hip_runtime.h>
#include <hip/hip_bf16.h>
#include <stdint.h>

// HVBiLSTM for MI455X (gfx1250): bf16 WMMA (v_wmma_f32_16x16x32_bf16) for both
// the fused input projection GEMM and the recurrent h@Wh GEMMs.
// B=32, HG=WG=24, D=768, HID=384. fp32 in/out, bf16 operands, f32 accumulate.

#define Bb 32
#define Dd 768
#define HID 384
#define G4 1536            // 4*HID
#define M_ALL 18432        // B*HG*WG
#define N_ALL 6144         // 4 directions * G4
#define Tt 24
#define SEQ 768            // B*24 sequences per direction

typedef __attribute__((ext_vector_type(16))) __bf16 v16bf;
typedef __attribute__((ext_vector_type(8)))  float  v8f;

union FragU { v16bf v; uint4 q[2]; };

__device__ __forceinline__ unsigned short f32_to_bf16(float f) {
  union { float f; uint32_t u; } cv; cv.f = f;
  uint32_t u = cv.u;
  uint32_t r = u + 0x7FFFu + ((u >> 16) & 1u);   // round-to-nearest-even
  return (unsigned short)(r >> 16);
}

// ---------------- pack kernels ----------------

// features fp32 -> bf16, same row-major layout [M_ALL][D]
__global__ __launch_bounds__(256) void pack_x(const float* __restrict__ feat,
                                              unsigned short* __restrict__ X) {
  size_t i = (size_t)blockIdx.x * 256 + threadIdx.x;   // 14,155,776 = multiple of 256
  X[i] = f32_to_bf16(feat[i]);
}

// WxP fragment layout: [kt(24)][nt(384)][lane(32)][i(16)] bf16.
// B-fragment per ISA: lane<16 -> col=lane, K = kt*32 + {0..7, 16..23};
//                     lane>=16 -> col=lane-16, K = kt*32 + {8..15, 24..31}.
__global__ __launch_bounds__(256) void pack_wx(const float* __restrict__ wx0, const float* __restrict__ wx1,
                                               const float* __restrict__ wx2, const float* __restrict__ wx3,
                                               const float* __restrict__ b0, const float* __restrict__ b1,
                                               const float* __restrict__ b2, const float* __restrict__ b3,
                                               unsigned short* __restrict__ WxP, float* __restrict__ ball) {
  size_t id = (size_t)blockIdx.x * 256 + threadIdx.x;  // 4,718,592 elements
  int i    = (int)(id & 15);
  int lane = (int)((id >> 4) & 31);
  size_t frag = id >> 9;                                // kt*384 + nt
  int nt = (int)(frag % 384);
  int kt = (int)(frag / 384);
  int n  = nt * 16 + (lane & 15);
  int k  = kt * 32 + ((lane < 16) ? 0 : 8) + ((i < 8) ? i : 16 + (i - 8));
  int dir = n / G4;
  int nc  = n % G4;
  const float* w = (dir == 0) ? wx0 : (dir == 1) ? wx1 : (dir == 2) ? wx2 : wx3;
  WxP[id] = f32_to_bf16(w[(size_t)k * G4 + nc]);
  if (id < N_ALL) {
    int d2 = (int)(id / G4), c2 = (int)(id % G4);
    const float* bb = (d2 == 0) ? b0 : (d2 == 1) ? b1 : (d2 == 2) ? b2 : b3;
    ball[id] = bb[c2];
  }
}

// WhP fragment layout: [dir(4)][kt(12)][nt(96)][lane(32)][i(16)] bf16.
__global__ __launch_bounds__(256) void pack_wh(const float* __restrict__ w0, const float* __restrict__ w1,
                                               const float* __restrict__ w2, const float* __restrict__ w3,
                                               unsigned short* __restrict__ WhP) {
  size_t id = (size_t)blockIdx.x * 256 + threadIdx.x;  // 2,359,296 elements
  int i    = (int)(id & 15);
  int lane = (int)((id >> 4) & 31);
  size_t frag = id >> 9;                                // dir*1152 + kt*96 + nt
  int nt  = (int)(frag % 96);
  int kt  = (int)((frag / 96) % 12);
  int dir = (int)(frag / (96 * 12));
  int n = nt * 16 + (lane & 15);
  int k = kt * 32 + ((lane < 16) ? 0 : 8) + ((i < 8) ? i : 16 + (i - 8));
  const float* w = (dir == 0) ? w0 : (dir == 1) ? w1 : (dir == 2) ? w2 : w3;
  WhP[id] = f32_to_bf16(w[(size_t)k * G4 + n]);
}

// ---------------- Z = X @ Wx_all + bias (fused 4-direction GEMM) ----------------
// Block: 256 threads (8 waves). Tile: M=32, N=512. Each wave: 2 mtiles x 4 ntiles.
__global__ __launch_bounds__(256) void zgemm(const unsigned short* __restrict__ X,
                                             const unsigned short* __restrict__ WxP,
                                             const float* __restrict__ ball,
                                             float* __restrict__ Z) {
  int wave = threadIdx.x >> 5;
  int lane = threadIdx.x & 31;
  int l15  = lane & 15;
  int kb   = (lane < 16) ? 0 : 8;      // A-fragment K base within 32-chunk
  int mbase = blockIdx.x * 32;
  int nbase = blockIdx.y * 512 + wave * 64;

  v8f acc[2][4];
#pragma unroll
  for (int mt = 0; mt < 2; ++mt)
#pragma unroll
    for (int nt = 0; nt < 4; ++nt) acc[mt][nt] = (v8f)0.0f;

  for (int kt = 0; kt < 24; ++kt) {
    FragU a[2];
#pragma unroll
    for (int mt = 0; mt < 2; ++mt) {
      const unsigned short* src = X + (size_t)(mbase + mt * 16 + l15) * Dd + kt * 32 + kb;
      a[mt].q[0] = *(const uint4*)(src);        // K = kb..kb+7
      a[mt].q[1] = *(const uint4*)(src + 16);   // K = kb+16..kb+23
    }
#pragma unroll
    for (int nt = 0; nt < 4; ++nt) {
      int ntile = (nbase >> 4) + nt;
      const unsigned short* bsrc = WxP + (((size_t)kt * 384 + ntile) * 32 + lane) * 16;
      FragU b;
      b.q[0] = *(const uint4*)(bsrc);
      b.q[1] = *(const uint4*)(bsrc + 8);
#pragma unroll
      for (int mt = 0; mt < 2; ++mt)
        acc[mt][nt] = __builtin_amdgcn_wmma_f32_16x16x32_bf16(
            false, a[mt].v, false, b.v, (short)0, acc[mt][nt], false, false);
    }
  }
#pragma unroll
  for (int nt = 0; nt < 4; ++nt) {
    int n = nbase + nt * 16 + l15;
    float bv = ball[n];
#pragma unroll
    for (int mt = 0; mt < 2; ++mt)
#pragma unroll
      for (int v = 0; v < 8; ++v) {
        int row = mbase + mt * 16 + v + 8 * (lane >> 4);
        Z[(size_t)row * N_ALL + n] = acc[mt][nt][v] + bv;
      }
  }
}

// ---------------- recurrent scan: one block = 16 sequences of one direction ----------------
// Per step: acc = z + h@Wh via WMMA (C operand seeds z), gates fully in registers
// (each wave owns matching i/f/cand/o column groups), h-state bf16 in LDS.
__global__ __launch_bounds__(256) void lstm_scan(const float* __restrict__ Z,
                                                 const unsigned short* __restrict__ WhP,
                                                 float* __restrict__ out) {
  __shared__ unsigned short h_lds[16 * HID];   // 12 KB

  int dir  = blockIdx.y;
  int sb   = blockIdx.x;            // 0..47
  int wave = threadIdx.x >> 5;
  int lane = threadIdx.x & 31;
  int l15  = lane & 15;
  int hi   = lane >> 4;

  for (int i = threadIdx.x; i < 16 * HID; i += 256) h_lds[i] = 0;
  __syncthreads();

  float c[3][8];
#pragma unroll
  for (int q = 0; q < 3; ++q)
#pragma unroll
    for (int v = 0; v < 8; ++v) c[q][v] = 0.0f;

  const bool rev  = (dir & 1);
  const bool vert = (dir >> 1);
  const int chanoff = dir * HID;

  int sB[8], sI[8];
#pragma unroll
  for (int v = 0; v < 8; ++v) {
    int s = sb * 16 + v + 8 * hi;
    sB[v] = s / 24; sI[v] = s % 24;
  }

  for (int tt = 0; tt < Tt; ++tt) {
    int t = rev ? (Tt - 1 - tt) : tt;
    size_t mg[8];
#pragma unroll
    for (int v = 0; v < 8; ++v)
      mg[v] = vert ? ((size_t)sB[v] * 576 + (size_t)t * 24 + sI[v])
                   : ((size_t)sB[v] * 576 + (size_t)sI[v] * 24 + t);

    // Seed accumulators with z (includes bias from zgemm).
    v8f acc[12];
#pragma unroll
    for (int p = 0; p < 12; ++p) {
      int g = p / 3, q = p % 3;
      int ntile = g * 24 + wave * 3 + q;
      int col = dir * G4 + ntile * 16 + l15;
      v8f tmpv;
#pragma unroll
      for (int v = 0; v < 8; ++v) tmpv[v] = Z[mg[v] * N_ALL + col];
      acc[p] = tmpv;
    }

    // g += h @ Wh : K = 384 in 12 steps of 32.
    for (int kt = 0; kt < 12; ++kt) {
      const unsigned short* asrc = h_lds + l15 * HID + kt * 32 + ((lane < 16) ? 0 : 8);
      FragU a;
      a.q[0] = *(const uint4*)(asrc);         // K = kb..kb+7
      a.q[1] = *(const uint4*)(asrc + 16);    // K = kb+16..kb+23
#pragma unroll
      for (int p = 0; p < 12; ++p) {
        int g = p / 3, q = p % 3;
        int ntile = g * 24 + wave * 3 + q;
        const unsigned short* bsrc = WhP + ((((size_t)dir * 12 + kt) * 96 + ntile) * 32 + lane) * 16;
        FragU b;
        b.q[0] = *(const uint4*)bsrc;
        b.q[1] = *(const uint4*)(bsrc + 8);
        acc[p] = __builtin_amdgcn_wmma_f32_16x16x32_bf16(
            false, a.v, false, b.v, (short)0, acc[p], false, false);
      }
    }
    __syncthreads();   // all h_lds reads for this step complete

    // Gates: wave w owns columns j in [w*48, w*48+48) across all four gate groups.
#pragma unroll
    for (int q = 0; q < 3; ++q) {
      int j0 = wave * 48 + q * 16 + l15;
#pragma unroll
      for (int v = 0; v < 8; ++v) {
        float gi = acc[q][v];
        float gf = acc[3 + q][v];
        float gc = acc[6 + q][v];
        float go = acc[9 + q][v];
        float iv = 1.0f / (1.0f + __expf(-gi));
        float fv = 1.0f / (1.0f + __expf(-gf));
        float cd = tanhf(gc);
        float ov = 1.0f / (1.0f + __expf(-go));
        float c2 = fv * c[q][v] + iv * cd;
        float h2 = ov * tanhf(c2);
        c[q][v] = c2;
        int m_local = v + 8 * hi;
        h_lds[m_local * HID + j0] = f32_to_bf16(h2);
        out[mg[v] * G4 + chanoff + j0] = h2;
      }
    }
    __syncthreads();   // h ready for next step
  }
}

// ---------------- launch ----------------

extern "C" void kernel_launch(void* const* d_in, const int* in_sizes, int n_in,
                              void* d_out, int out_size, void* d_ws, size_t ws_size,
                              hipStream_t stream) {
  (void)in_sizes; (void)n_in; (void)out_size; (void)ws_size;
  const float* feat = (const float*)d_in[0];
  const float* hfWx = (const float*)d_in[1];
  const float* hfWh = (const float*)d_in[2];
  const float* hfb  = (const float*)d_in[3];
  const float* hbWx = (const float*)d_in[4];
  const float* hbWh = (const float*)d_in[5];
  const float* hbb  = (const float*)d_in[6];
  const float* vfWx = (const float*)d_in[7];
  const float* vfWh = (const float*)d_in[8];
  const float* vfb  = (const float*)d_in[9];
  const float* vbWx = (const float*)d_in[10];
  const float* vbWh = (const float*)d_in[11];
  const float* vbb  = (const float*)d_in[12];
  float* out = (float*)d_out;

  char* ws = (char*)d_ws;
  // workspace layout (all 16B aligned):
  float*          Z    = (float*)(ws);                                   // 452,984,832 B
  unsigned short* X    = (unsigned short*)(ws + 452984832ull);           //  28,311,552 B
  unsigned short* WxP  = (unsigned short*)(ws + 481296384ull);           //   9,437,184 B
  unsigned short* WhP  = (unsigned short*)(ws + 490733568ull);           //   4,718,592 B
  float*          ball = (float*)(ws + 495452160ull);                    //      24,576 B

  pack_x <<<55296, 256, 0, stream>>>(feat, X);
  pack_wx<<<18432, 256, 0, stream>>>(hfWx, hbWx, vfWx, vbWx, hfb, hbb, vfb, vbb, WxP, ball);
  pack_wh<<< 9216, 256, 0, stream>>>(hfWh, hbWh, vfWh, vbWh, WhP);
  zgemm  <<<dim3(M_ALL / 32, N_ALL / 512), 256, 0, stream>>>(X, WxP, ball, Z);
  lstm_scan<<<dim3(SEQ / 16, 4), 256, 0, stream>>>(Z, WhP, out);
}
